// TFAdaptiveSoftmaxMask_73134703116926
// MI455X (gfx1250) — matrix operand order repaired and checked by reading, not compiled, not run on hardware
//
#include <hip/hip_runtime.h>

typedef __attribute__((ext_vector_type(16))) _Float16 v16h;
typedef __attribute__((ext_vector_type(8)))  float    v8f;
typedef __attribute__((ext_vector_type(4)))  unsigned int u32x4;
typedef __attribute__((ext_vector_type(4)))  float    f32x4;

#define VOCAB 267735
#define NTOK  512
#define DEMB  1024
#define BM    64     // rows per block (shared by all 4 waves)
#define BN    128    // cols per block (32 per wave)
#define MF    4      // 4 row fragments per wave
#define NF    2      // 2 col fragments per wave

// ---------------- fragment loaders (CDNA5 wave32 WMMA layouts) ----------------
union AFragU { v16h h; u32x4 u[2]; };

// A (16x32 f16, MxK): lane holds row M=lane%16; K runs [base..base+15],
// base = (lane>=16)?16:0 -> two unconditional 16B loads.
__device__ __forceinline__ v16h load_a_full(const _Float16* __restrict__ A,
                                            int row, int d, int k0, int lane) {
  AFragU fr;
  const _Float16* p = A + (size_t)row * d + k0 + ((lane & 16) ? 16 : 0);
  fr.u[0] = *(const u32x4*)(p);
  fr.u[1] = *(const u32x4*)(p + 8);
  return fr.h;
}

// d == 16 (tail-3): K=16..31 zero-padded
__device__ __forceinline__ v16h load_a_pad16(const _Float16* __restrict__ A,
                                             int row, int lane) {
  AFragU fr;
  const _Float16* p = A + (size_t)row * 16;
  fr.u[0] = (lane & 16) ? u32x4{} : *(const u32x4*)(p);
  fr.u[1] = (lane & 16) ? u32x4{} : *(const u32x4*)(p + 8);
  return fr.h;
}

__device__ __forceinline__ v16h cvt16(f32x4 q0, f32x4 q1, f32x4 q2, f32x4 q3) {
  v16h b;
#pragma unroll
  for (int t = 0; t < 4; ++t) {
    b[t]      = (_Float16)q0[t];
    b[4 + t]  = (_Float16)q1[t];
    b[8 + t]  = (_Float16)q2[t];
    b[12 + t] = (_Float16)q3[t];
  }
  return b;
}

// B (32x16 f16, KxN): lane holds column N=lane%16; K = (lane>=16?16:0)+0..15,
// i.e. 16 contiguous fp32 of one W row (W row-major [n][d], B = W^T).
__device__ __forceinline__ v16h load_b_full(const float* __restrict__ Wm,
                                            int d, int col, int k0, int lane) {
  const float* p = Wm + (size_t)col * d + k0 + ((lane & 16) ? 16 : 0);
  return cvt16(*(const f32x4*)(p), *(const f32x4*)(p + 4),
               *(const f32x4*)(p + 8), *(const f32x4*)(p + 12));
}

__device__ __forceinline__ v16h load_b_pad16(const float* __restrict__ Wm,
                                             int col, int lane) {
  const float* p = Wm + (size_t)col * 16;
  v16h b = cvt16(*(const f32x4*)(p), *(const f32x4*)(p + 4),
                 *(const f32x4*)(p + 8), *(const f32x4*)(p + 12));
#pragma unroll
  for (int t = 0; t < 16; ++t) b[t] = (lane & 16) ? (_Float16)0.f : b[t];
  return b;
}

// wave tile 64x32 = 4x2 v_wmma_f32_16x16x32_f16 per K-step of 32
__device__ __forceinline__ void gemm_core(const _Float16* __restrict__ A,
                                          const float* __restrict__ Wm,
                                          int n, int d, int rBase, int cBase,
                                          int lane, v8f acc[MF][NF]) {
  const int laneM = lane & 15;
  int cols[NF];
#pragma unroll
  for (int nn = 0; nn < NF; ++nn) {
    int c = cBase + nn * 16 + laneM;
    cols[nn] = (c < n) ? c : (n - 1);   // clamp; epilogue guards discard extras
  }
  if (d >= 32) {                        // uniform branch: d in {1024,256,64}
    for (int k0 = 0; k0 < d; k0 += 32) {
      v16h a[MF], b[NF];
#pragma unroll
      for (int m = 0; m < MF; ++m)
        a[m] = load_a_full(A, rBase + m * 16 + laneM, d, k0, lane);
#pragma unroll
      for (int nn = 0; nn < NF; ++nn)
        b[nn] = load_b_full(Wm, d, cols[nn], k0, lane);
#pragma unroll
      for (int m = 0; m < MF; ++m)
#pragma unroll
        for (int nn = 0; nn < NF; ++nn)
          acc[m][nn] = __builtin_amdgcn_wmma_f32_16x16x32_f16(
              false, a[m], false, b[nn], (short)0, acc[m][nn], false, false);
    }
  } else {                              // d == 16: one zero-padded K-step
    v16h a[MF], b[NF];
#pragma unroll
    for (int m = 0; m < MF; ++m)
      a[m] = load_a_pad16(A, rBase + m * 16 + laneM, lane);
#pragma unroll
    for (int nn = 0; nn < NF; ++nn)
      b[nn] = load_b_pad16(Wm, cols[nn], lane);
#pragma unroll
    for (int m = 0; m < MF; ++m)
#pragma unroll
      for (int nn = 0; nn < NF; ++nn)
        acc[m][nn] = __builtin_amdgcn_wmma_f32_16x16x32_f16(
            false, a[m], false, b[nn], (short)0, acc[m][nn], false, false);
  }
}

// MODE 0: per-block-column partial sum-of-exp (deterministic, no atomics)
// MODE 1: recompute logits (bitwise-identical), write final log-probs to out
template <int MODE>
__global__ void adasoft_gemm_kernel(const _Float16* __restrict__ A,
                                    const float* __restrict__ Wm,
                                    const float* __restrict__ bias,
                                    int n, int d,
                                    float* __restrict__ partials, int pstride,
                                    const float* __restrict__ logS,
                                    const float* __restrict__ addv,
                                    float* __restrict__ outp, int nWrite) {
  const int lane  = threadIdx.x & 31;
  const int wave  = threadIdx.x >> 5;           // 0..3 -> column slab
  const int rBase = blockIdx.y * BM;
  const int cBase = blockIdx.x * BN + wave * 32;
  const int laneM = lane & 15;
  const int rHi   = (lane >> 4) << 3;           // C/D: lanes 16..31 hold rows M+8

  v8f acc[MF][NF] = {};
  gemm_core(A, Wm, n, d, rBase, cBase, lane, acc);

  __shared__ float part[4][BM];

  if (MODE == 0) {
    bool  ok[NF];
    float bc[NF];
#pragma unroll
    for (int nn = 0; nn < NF; ++nn) {
      int col = cBase + nn * 16 + laneM;
      ok[nn] = (col < n);
      bc[nn] = bias[ok[nn] ? col : (n - 1)];
    }
#pragma unroll
    for (int m = 0; m < MF; ++m) {
#pragma unroll
      for (int j = 0; j < 8; ++j) {
        float v = 0.f;
#pragma unroll
        for (int nn = 0; nn < NF; ++nn)
          v += ok[nn] ? __expf(acc[m][nn][j] + bc[nn]) : 0.f;   // cndmask, no branch
#pragma unroll
        for (int s = 1; s < 16; s <<= 1) v += __shfl_xor(v, s, 32);  // 16-lane halves
        if (laneM == 0) part[wave][m * 16 + rHi + j] = v;
      }
    }
    __syncthreads();
    if (threadIdx.x < BM) {
      int r = threadIdx.x;
      float s = part[0][r] + part[1][r] + part[2][r] + part[3][r];
      partials[(size_t)(rBase + r) * pstride + blockIdx.x] = s;
    }
  } else {
#pragma unroll
    for (int nn = 0; nn < NF; ++nn) {
      int col = cBase + nn * 16 + laneM;
      if (col < nWrite) {
        float bc = bias[col];
#pragma unroll
        for (int m = 0; m < MF; ++m)
#pragma unroll
          for (int j = 0; j < 8; ++j) {
            int row = rBase + m * 16 + rHi + j;
            float v = acc[m][nn][j] + bc - logS[row];
            if (addv) v += addv[row];
            outp[(size_t)row * VOCAB + col] = v;
          }
      }
    }
  }
}

// Y_i = H @ P_i^T, output f16
__global__ void adasoft_proj_kernel(const _Float16* __restrict__ A,
                                    const float* __restrict__ P,
                                    _Float16* __restrict__ Y, int N, int d) {
  const int lane  = threadIdx.x & 31;
  const int wave  = threadIdx.x >> 5;
  const int rBase = blockIdx.y * BM;
  const int cBase = blockIdx.x * BN + wave * 32;
  const int laneM = lane & 15;
  const int rHi   = (lane >> 4) << 3;
  v8f acc[MF][NF] = {};
  gemm_core(A, P, N, d, rBase, cBase, lane, acc);
#pragma unroll
  for (int nn = 0; nn < NF; ++nn) {
    int col = cBase + nn * 16 + laneM;
    if (col < N)
#pragma unroll
      for (int m = 0; m < MF; ++m)
#pragma unroll
        for (int j = 0; j < 8; ++j) {
          int row = rBase + m * 16 + rHi + j;
          Y[(size_t)row * N + col] = (_Float16)acc[m][nn][j];
        }
  }
}

__global__ void adasoft_conv_kernel(const float* __restrict__ h,
                                    _Float16* __restrict__ o, int n) {
  int i = blockIdx.x * blockDim.x + threadIdx.x;
  if (i < n) o[i] = (_Float16)h[i];
}

// 3 cluster-logit columns of the head + their exp-sum into the extra partial slot
__global__ void adasoft_cluster_kernel(const _Float16* __restrict__ Y0,
                                       const float* __restrict__ cw,
                                       const float* __restrict__ cb,
                                       float* __restrict__ clog,
                                       float* __restrict__ part0, int pstride0, int slot) {
  int warp = (blockIdx.x * blockDim.x + threadIdx.x) >> 5;
  int lane = threadIdx.x & 31;
  if (warp >= NTOK) return;
  const _Float16* y = Y0 + (size_t)warp * DEMB;
  float cl[3];
#pragma unroll
  for (int j = 0; j < 3; ++j) {
    float s = 0.f;
    for (int k = lane; k < DEMB; k += 32) s += (float)y[k] * cw[j * DEMB + k];
    for (int sh = 16; sh; sh >>= 1) s += __shfl_xor(s, sh, 32);
    cl[j] = s + cb[j];
  }
  if (lane == 0) {
    clog[warp * 3 + 0] = cl[0];
    clog[warp * 3 + 1] = cl[1];
    clog[warp * 3 + 2] = cl[2];
    part0[(size_t)warp * pstride0 + slot] = __expf(cl[0]) + __expf(cl[1]) + __expf(cl[2]);
  }
}

struct RedArgs { const float* p[4]; int stride[4]; };

__global__ void adasoft_reduce_kernel(RedArgs ra, float* __restrict__ logS) {
  int g    = (blockIdx.x * blockDim.x + threadIdx.x) >> 5;  // one warp per (cluster,row)
  int lane = threadIdx.x & 31;
  if (g >= 4 * NTOK) return;
  int ci = g >> 9, row = g & (NTOK - 1);
  const float* p = ra.p[ci] + (size_t)row * ra.stride[ci];
  int cnt = ra.stride[ci];
  float s = 0.f;
  for (int k = lane; k < cnt; k += 32) s += p[k];
  for (int sh = 16; sh; sh >>= 1) s += __shfl_xor(s, sh, 32);
  if (lane == 0) logS[ci * NTOK + row] = logf(s);
}

__global__ void adasoft_clusterlp_kernel(const float* __restrict__ clog,
                                         const float* __restrict__ logS,
                                         float* __restrict__ clp) {
  int t = blockIdx.x * blockDim.x + threadIdx.x;
  if (t >= 3 * NTOK) return;
  int j = t / NTOK, row = t % NTOK;
  clp[j * NTOK + row] = clog[row * 3 + j] - logS[row];  // logS cluster 0 = head
}

__global__ void adasoft_loss_kernel(const float* __restrict__ out,
                                    const int* __restrict__ target,
                                    float* __restrict__ lossOut) {
  __shared__ float sh[NTOK];
  int t  = threadIdx.x;
  int tg = target[t];
  float lp = 0.f;
  if (tg >= 0 && tg < VOCAB) lp = out[(size_t)t * VOCAB + tg];
  sh[t] = -lp;
  __syncthreads();
  for (int s = NTOK / 2; s; s >>= 1) {
    if (t < s) sh[t] += sh[t + s];
    __syncthreads();
  }
  if (t == 0) *lossOut = sh[0] / (float)NTOK;
}

extern "C" void kernel_launch(void* const* d_in, const int* in_sizes, int n_in,
                              void* d_out, int out_size, void* d_ws, size_t ws_size,
                              hipStream_t stream) {
  (void)in_sizes; (void)n_in; (void)out_size; (void)ws_size;
  const float* hidden = (const float*)d_in[0];
  const int*   target = (const int*)d_in[1];
  const float* cw     = (const float*)d_in[2];
  const float* cb     = (const float*)d_in[3];
  const float* Wm[4]  = {(const float*)d_in[4], (const float*)d_in[7], (const float*)d_in[10], (const float*)d_in[13]};
  const float* bs[4]  = {(const float*)d_in[5], (const float*)d_in[8], (const float*)d_in[11], (const float*)d_in[14]};
  const float* Pm[4]  = {(const float*)d_in[6], (const float*)d_in[9], (const float*)d_in[12], (const float*)d_in[15]};

  static const int nCls[4]   = {20000, 20000, 160000, 67735};
  static const int dPrj[4]   = {1024, 256, 64, 16};
  static const int colOff[4] = {0, 20000, 40000, 200000};

  float* out     = (float*)d_out;
  float* lossOut = out + (size_t)NTOK * VOCAB;

  // ---- workspace carve-out (~7 MB) ----
  char* w = (char*)d_ws;
  _Float16* H16 = (_Float16*)w; w += (size_t)NTOK * DEMB * 2;
  _Float16* Y[4];
  for (int i = 0; i < 4; ++i) { Y[i] = (_Float16*)w; w += (size_t)NTOK * dPrj[i] * 2; }
  float* clog = (float*)w; w += (size_t)NTOK * 3 * 4;
  float* logS = (float*)w; w += (size_t)4 * NTOK * 4;
  float* clp  = (float*)w; w += (size_t)3 * NTOK * 4;
  float* part[4]; int pstr[4];
  for (int i = 0; i < 4; ++i) {
    int gx = (nCls[i] + BN - 1) / BN;
    pstr[i] = gx + (i == 0 ? 1 : 0);              // head: +1 slot for cluster cols
    part[i] = (float*)w; w += (size_t)NTOK * pstr[i] * 4;
  }

  // 1. hidden f32 -> f16
  adasoft_conv_kernel<<<(NTOK * DEMB + 255) / 256, 256, 0, stream>>>(hidden, H16, NTOK * DEMB);

  // 2. projections Y_i = H @ P_i^T   (WMMA f16)
  for (int i = 0; i < 4; ++i) {
    dim3 g((dPrj[i] + BN - 1) / BN, NTOK / BM);
    adasoft_proj_kernel<<<g, 128, 0, stream>>>(H16, Pm[i], Y[i], dPrj[i], DEMB);
  }

  // 3. pass A: GEMM + per-tile partial sum-of-exp (logits stay in registers)
  for (int i = 0; i < 4; ++i) {
    dim3 g((nCls[i] + BN - 1) / BN, NTOK / BM);
    adasoft_gemm_kernel<0><<<g, 128, 0, stream>>>(Y[i], Wm[i], bs[i], nCls[i], dPrj[i],
                                                  part[i], pstr[i], nullptr, nullptr, nullptr, 0);
  }

  // 4. head cluster columns (3) + their exp-sum partial
  adasoft_cluster_kernel<<<(NTOK * 32 + 255) / 256, 256, 0, stream>>>(
      Y[0], cw, cb, clog, part[0], pstr[0], pstr[0] - 1);

  // 5. reduce partials -> log(sum exp) per row per cluster (deterministic)
  RedArgs ra;
  for (int i = 0; i < 4; ++i) { ra.p[i] = part[i]; ra.stride[i] = pstr[i]; }
  adasoft_reduce_kernel<<<(4 * NTOK * 32 + 255) / 256, 256, 0, stream>>>(ra, logS);

  // 6. cluster log-probs for the tails
  adasoft_clusterlp_kernel<<<(3 * NTOK + 255) / 256, 256, 0, stream>>>(clog, logS, clp);

  // 7. pass B: recompute GEMM (bitwise-identical), write final log-probs
  for (int i = 0; i < 4; ++i) {
    dim3 g((nCls[i] + BN - 1) / BN, NTOK / BM);
    adasoft_gemm_kernel<1><<<g, 128, 0, stream>>>(Y[i], Wm[i], bs[i], nCls[i], dPrj[i],
                                                  nullptr, 0, logS + i * NTOK,
                                                  (i == 0) ? nullptr : (clp + (i - 1) * NTOK),
                                                  out + colOff[i], nCls[i]);
  }

  // 8. loss = mean(-lp[target])
  adasoft_loss_kernel<<<1, NTOK, 0, stream>>>(out, target, lossOut);
}
